// CTRNNCell_67413806678557
// MI455X (gfx1250) — matrix operand order repaired and checked by reading, not compiled, not run on hardware
//
#include <hip/hip_runtime.h>

// ---------------------------------------------------------------------------
// CTRNN RK4 unroll for MI455X (gfx1250), bf16 WMMA with f32 accumulation.
//
// Roofline: 24 recurrent GEMMs of 32768x512x512 => ~412 GFLOP, I/O ~160MB
// (~7us @ 23.3 TB/s) => compute bound => v_wmma_f32_16x16x32_bf16 (K=32 vs
// K=4 for fp32 WMMA). x@K is evaluation-invariant (computed once); batch rows
// are independent so each wave runs the whole 24-GEMM recurrence on its 16
// rows with no global sync. Packed bf16 weights (768KB) stay L2-resident.
//
// v2: 2 LDS buffers/wave instead of 3 (h master kept f32 in workspace) =>
// LDS 133,120 B/WG => 2 WGs/WGP => 2 waves/SIMD to hide the L2 B-fragment
// latency the v1 disasm showed as s_wait_loadcnt stalls before each wmma.
// Workspace use: 1MB packed weights + 3 x 64MB f32 (xk, hacc, hws) = 193MB.
// ---------------------------------------------------------------------------

typedef __attribute__((ext_vector_type(16))) __bf16 v16bf;
typedef __attribute__((ext_vector_type(8)))  float  v8f;

#define UNITS  512
#define D_IN   256
#define NUNF   6
#define LSTR   520            // LDS row stride (bf16): 1040B rows -> 16B aligned,
                              // 260 dwords % 64 banks = 4 -> conflict-free A reads
#define WAVES  4              // waves per workgroup
#define ROWS_PER_WG (WAVES*16)

// ---- fragment loaders ------------------------------------------------------

// A fragment (16x32 bf16, row-major source in LDS).
__device__ __forceinline__ v16bf load_a(const __bf16* buf, int kc, int lr, int hi) {
  const __bf16* p = buf + lr * LSTR + kc * 32 + hi * 16;
  union { v16bf v; uint4 u[2]; } A;
  A.u[0] = *(const uint4*)(p);
  A.u[1] = *(const uint4*)(p + 8);
  return A.v;
}

// B fragment from pre-packed global weights: frag = 32 lanes x 32 bytes.
__device__ __forceinline__ v16bf load_b(const __bf16* pack, int frag, int lane) {
  const uint4* p = (const uint4*)pack + ((size_t)frag * 32 + lane) * 2;
  union { v16bf v; uint4 u[2]; } Bf;
  Bf.u[0] = p[0];
  Bf.u[1] = p[1];
  return Bf.v;
}

#define WMMA_BF16(a, b, c) \
  __builtin_amdgcn_wmma_f32_16x16x32_bf16(false, (a), false, (b), (short)0, (c), false, false)

// ---------------------------------------------------------------------------
// Pack K (256x512 f32) and R (512x512 f32) into bf16 B-fragment layout:
// fragment f covers N-tile nt, K-chunk kc; lane l holds column nt*16+(l&15),
// 16 contiguous K values starting at kc*32 + (l>>4)*16.
// ---------------------------------------------------------------------------
__global__ void pack_weights(const float* __restrict__ Kw,
                             const float* __restrict__ Rw,
                             __bf16* __restrict__ Kp,
                             __bf16* __restrict__ Rp) {
  int idx = blockIdx.x * blockDim.x + threadIdx.x;
  if (idx < 256 * 32) {                       // K: 32 nt * 8 kc fragments
    int l = idx & 31, f = idx >> 5;
    int kc = f & 7, nt = f >> 3;
    int n  = nt * 16 + (l & 15);
    int kb = kc * 32 + (l >> 4) * 16;
#pragma unroll
    for (int j = 0; j < 16; ++j)
      Kp[(size_t)idx * 16 + j] = (__bf16)Kw[(size_t)(kb + j) * UNITS + n];
  } else if (idx < 256 * 32 + 512 * 32) {     // R: 32 nt * 16 kc fragments
    int id2 = idx - 256 * 32;
    int l = id2 & 31, f = id2 >> 5;
    int kc = f & 15, nt = f >> 4;
    int n  = nt * 16 + (l & 15);
    int kb = kc * 32 + (l >> 4) * 16;
#pragma unroll
    for (int j = 0; j < 16; ++j)
      Rp[(size_t)id2 * 16 + j] = (__bf16)Rw[(size_t)(kb + j) * UNITS + n];
  }
}

// ---------------------------------------------------------------------------
// Main kernel: each wave owns 16 batch rows, runs 6 RK4 unfolds in place.
// ---------------------------------------------------------------------------
__global__ void __launch_bounds__(128, 2)
ctrnn_kernel(const float* __restrict__ x,  const float* __restrict__ h0,
             const float* __restrict__ bias, const float* __restrict__ scale,
             const __bf16* __restrict__ Kp, const __bf16* __restrict__ Rp,
             float* __restrict__ xk, float* __restrict__ hacc,
             float* __restrict__ hws, float* __restrict__ out) {
  __shared__ __align__(16) __bf16 smem[WAVES * 2 * 16 * LSTR]; // 133,120 B

  const int tid  = threadIdx.x;
  const int lane = tid & 31;
  const int wave = tid >> 5;
  const int lr   = lane & 15;   // C-layout: column-in-tile / A-layout: row
  const int hi   = lane >> 4;
  const int rowbase = (blockIdx.x * WAVES + wave) * 16;

  __bf16* eA = smem + (wave * 2 + 0) * 16 * LSTR;  // ping (holds h bf16 at s0)
  __bf16* eB = smem + (wave * 2 + 1) * 16 * LSTR;  // pong (holds x at prologue)

  // ---- phase 0: stage x (into eB, cols 0..255) and h0 bf16 (into eA) -----
  for (int i = lane; i < 16 * D_IN; i += 32) {
    int r = i >> 8, c = i & (D_IN - 1);
    eB[r * LSTR + c] = (__bf16)x[(size_t)(rowbase + r) * D_IN + c];
  }
  for (int i = lane; i < 16 * UNITS; i += 32) {
    int r = i >> 9, c = i & (UNITS - 1);
    eA[r * LSTR + c] = (__bf16)h0[(size_t)(rowbase + r) * UNITS + c];
  }
  __syncthreads();

  // ---- phase 1: XK = x@K + bias (computed once, reused by all 24 evals) --
  for (int nt = 0; nt < 32; nt += 2) {
    v8f a0 = {}, a1 = {};
#pragma unroll
    for (int kc = 0; kc < 8; ++kc) {
      v16bf af = load_a(eB, kc, lr, hi);
      a0 = WMMA_BF16(af, load_b(Kp, (nt + 0) * 8 + kc, lane), a0);
      a1 = WMMA_BF16(af, load_b(Kp, (nt + 1) * 8 + kc, lane), a1);
    }
    int c0 = nt * 16 + lr, c1 = (nt + 1) * 16 + lr;
    float b0 = bias[c0], b1 = bias[c1];
#pragma unroll
    for (int r = 0; r < 8; ++r) {
      size_t row = (size_t)(rowbase + hi * 8 + r);
      xk[row * UNITS + c0] = a0[r] + b0;
      xk[row * UNITS + c1] = a1[r] + b1;
    }
  }
  __syncthreads();   // eB is reused as an eval buffer below

  // ---- phase 2: 6 RK4 unfolds, 4 stages each -----------------------------
  const float dt  = 1.0f / (float)NUNF;
  const float c16 = dt / 6.0f, c13 = dt / 3.0f, chf = dt * 0.5f;

  for (int u = 0; u < NUNF; ++u) {
    const float* hsrc = (u == 0) ? h0 : hws;   // f32 master copy of h
#pragma unroll 1
    for (int s = 0; s < 4; ++s) {
      // ping-pong: s0 eA->eB, s1 eB->eA, s2 eA->eB, s3 eB->eA(h_next)
      const __bf16* src = (s & 1) ? eB : eA;
      __bf16*       dst = (s & 1) ? eA : eB;
      const float   evc = (s == 2) ? dt : chf;
      const float   hcc = (s == 0 || s == 3) ? c16 : c13;

      for (int nt = 0; nt < 32; nt += 2) {
        v8f acc0 = {}, acc1 = {};
#pragma unroll
        for (int kc = 0; kc < 16; ++kc) {
          v16bf af = load_a(src, kc, lr, hi);
          acc0 = WMMA_BF16(af, load_b(Rp, (nt + 0) * 16 + kc, lane), acc0);
          acc1 = WMMA_BF16(af, load_b(Rp, (nt + 1) * 16 + kc, lane), acc1);
        }
#pragma unroll
        for (int t = 0; t < 2; ++t) {
          v8f acc = t ? acc1 : acc0;
          int col = (nt + t) * 16 + lr;
          float sc = scale[col];
#pragma unroll
          for (int r = 0; r < 8; ++r) {
            int lrow = hi * 8 + r;
            size_t gi = (size_t)(rowbase + lrow) * UNITS + col;
            float pre = xk[gi] + acc[r];
            float hev = (float)src[lrow * LSTR + col];   // h_eval this stage
            float kv  = sc * tanhf(pre) - hev;           // dfdt (TAU = 1)
            if (s == 0) {
              float hv = hsrc[gi];
              hacc[gi] = hv + c16 * kv;                  // start RK4 sum
              dst[lrow * LSTR + col] = (__bf16)(hv + chf * kv);
            } else if (s < 3) {
              float hv = hsrc[gi];
              hacc[gi] += c13 * kv;
              dst[lrow * LSTR + col] = (__bf16)(hv + evc * kv);
            } else {
              float hn = hacc[gi] + c16 * kv;            // h_{u+1}
              dst[lrow * LSTR + col] = (__bf16)hn;       // s0 src next unfold
              if (u == NUNF - 1) out[gi] = hn;
              else               hws[gi] = hn;
            }
          }
        }
      }
      __syncthreads();  // C-layout writes -> A-layout reads next stage
    }
  }
}

// ---------------------------------------------------------------------------
extern "C" void kernel_launch(void* const* d_in, const int* in_sizes, int n_in,
                              void* d_out, int out_size, void* d_ws, size_t ws_size,
                              hipStream_t stream) {
  const float* x     = (const float*)d_in[0];  // 32768 x 256
  const float* h0    = (const float*)d_in[1];  // 32768 x 512
  const float* Kw    = (const float*)d_in[2];  // 256 x 512
  const float* Rw    = (const float*)d_in[3];  // 512 x 512
  const float* bias  = (const float*)d_in[4];  // 512
  const float* scale = (const float*)d_in[5];  // 512

  char* ws = (char*)d_ws;
  __bf16* Kp   = (__bf16*)ws;                                     // 256 KB
  __bf16* Rp   = (__bf16*)(ws + 256 * 32 * 16 * 2);               // 512 KB
  float*  xk   = (float*)(ws + ((size_t)1 << 20));                // 64 MB
  float*  hacc = (float*)(ws + ((size_t)1 << 20) + ((size_t)64 << 20));   // 64 MB
  float*  hws  = (float*)(ws + ((size_t)1 << 20) + ((size_t)128 << 20));  // 64 MB

  pack_weights<<<96, 256, 0, stream>>>(Kw, Rw, Kp, Rp);
  ctrnn_kernel<<<32768 / ROWS_PER_WG, WAVES * 32, 0, stream>>>(
      x, h0, bias, scale, Kp, Rp, xk, hacc, hws, (float*)d_out);
}
